// LorentzAssignment_52123723104445
// MI455X (gfx1250) — matrix operand order repaired and checked by reading, not compiled, not run on hardware
//
#include <hip/hip_runtime.h>
#include <math.h>

typedef __bf16 bf16_t;
typedef __attribute__((ext_vector_type(16))) __bf16 v16bf;
typedef __attribute__((ext_vector_type(8)))  __bf16 v8bf;
typedef __attribute__((ext_vector_type(4)))  __bf16 v4bf;
typedef __attribute__((ext_vector_type(8)))  float  v8f;

#define DHID 128
#define EPSV 1e-8f
#define INV_TAU 5.0f   // 1/0.2

// ---------------------------------------------------------------- helpers
__device__ __forceinline__ unsigned encodeOrd(float f) {
  unsigned b = __float_as_uint(f);
  return (b & 0x80000000u) ? ~b : (b | 0x80000000u);
}
__device__ __forceinline__ float decodeOrd(unsigned u) {
  unsigned b = (u & 0x80000000u) ? (u & 0x7FFFFFFFu) : ~u;
  return __uint_as_float(b);
}

// JAX threefry2x32 (20 rounds), key = (0, seed)
__device__ __forceinline__ void threefry2x32(unsigned k0, unsigned k1,
                                             unsigned x0, unsigned x1,
                                             unsigned& o0, unsigned& o1) {
  const unsigned ks2 = k0 ^ k1 ^ 0x1BD11BDAu;
  unsigned ks[3] = {k0, k1, ks2};
  x0 += ks[0]; x1 += ks[1];
  const unsigned rotA[4] = {13u, 15u, 26u, 6u};
  const unsigned rotB[4] = {17u, 29u, 16u, 24u};
#pragma unroll
  for (int g = 0; g < 5; ++g) {
    const unsigned* R = (g & 1) ? rotB : rotA;
#pragma unroll
    for (int r = 0; r < 4; ++r) {
      x0 += x1;
      x1 = (x1 << R[r]) | (x1 >> (32u - R[r]));
      x1 ^= x0;
    }
    x0 += ks[(g + 1) % 3];
    x1 += ks[(g + 2) % 3] + (unsigned)(g + 1);
  }
  o0 = x0; o1 = x1;
}

// ------------------------------------------------------- WMMA frag loads
// A-frag (16x32 bf16): ISA 16-bit A layout. lane&15 = row; lanes 0-15 hold
// K k0..k0+7 (v0-3) and k0+16..k0+23 (v4-7); lanes 16-31 are shifted by +8.
__device__ __forceinline__ v16bf load_a_frag(const bf16_t* X, int N, int m0,
                                             int kc, int lane) {
  int r = m0 + (lane & 15);
  if (r > N - 1) r = N - 1;                 // clamp: keep EXEC all-ones
  int k0 = kc + ((lane >> 4) << 3);
  const bf16_t* p = X + (size_t)r * DHID + k0;
  v8bf lo = *(const v8bf*)(p);
  v8bf hi = *(const v8bf*)(p + 16);
  return __builtin_shufflevector(lo, hi, 0,1,2,3,4,5,6,7,8,9,10,11,12,13,14,15);
}

// B-frag (32x16 bf16, B^T = W row-major): lane&15 = output column n;
// lanes 0-15 hold K kc..kc+7 and kc+16..kc+23; lanes 16-31 are shifted by +8.
__device__ __forceinline__ v16bf load_b_frag(const bf16_t* W, int n0,
                                             int kc, int lane) {
  int n = n0 + (lane & 15);
  int k0 = kc + ((lane >> 4) << 3);
  const bf16_t* p = W + (size_t)n * DHID + k0;
  v8bf lo = *(const v8bf*)(p);
  v8bf hi = *(const v8bf*)(p + 16);
  return __builtin_shufflevector(lo, hi, 0,1,2,3,4,5,6,7,8,9,10,11,12,13,14,15);
}

// ------------------------------------------------------------- kernels
__global__ __launch_bounds__(256) void convert_bf16_kernel(const float* s,
                                                           bf16_t* d, int n) {
  int i = blockIdx.x * 256 + threadIdx.x;
  if (i < n) d[i] = (bf16_t)s[i];
}

// per-node: x -> bf16, logmap0(x) -> bf16.  one wave per node row.
__global__ __launch_bounds__(256) void prep_nodes_kernel(const float* x,
                                                         bf16_t* Xb, bf16_t* Lb,
                                                         int N) {
  const int lane = threadIdx.x & 31, wave = threadIdx.x >> 5;
  const int row = blockIdx.x * 8 + wave;
  if (row >= N) return;
  const float4 v = ((const float4*)(x + (size_t)row * DHID))[lane];
  float d4 = v.x * v.x + v.y * v.y + v.z * v.z + v.w * v.w;
  float part = (lane == 0) ? (d4 - v.x * v.x) : d4;   // exclude time coord
  for (int off = 16; off >= 1; off >>= 1) part += __shfl_xor(part, off, 32);
  const float x0  = __shfl(v.x, 0, 32);
  const float nrm = fmaxf(sqrtf(part), EPSV);
  const float xc  = fmaxf(x0, 1.0f + 1e-6f);
  const float dgeo = logf(xc + sqrtf(xc * xc - 1.0f));  // arccosh
  const float r = dgeo / nrm;
  float4 lm = make_float4(v.x * r, v.y * r, v.z * r, v.w * r);
  if (lane == 0) lm.x = 0.0f;
  v4bf xb; xb[0] = (bf16_t)v.x;  xb[1] = (bf16_t)v.y;  xb[2] = (bf16_t)v.z;  xb[3] = (bf16_t)v.w;
  v4bf lb; lb[0] = (bf16_t)lm.x; lb[1] = (bf16_t)lm.y; lb[2] = (bf16_t)lm.z; lb[3] = (bf16_t)lm.w;
  *(v4bf*)(Xb + (size_t)row * DHID + lane * 4) = xb;
  *(v4bf*)(Lb + (size_t)row * DHID + lane * 4) = lb;
}

// y = Lb @ Wa^T, then row softmax -> ass.  Block = 16 rows x 128 cols.
__global__ __launch_bounds__(256) void gemm_softmax_kernel(const bf16_t* Lb,
                                                           const bf16_t* Wab,
                                                           float* ass, int N) {
  __shared__ float ytile[16][DHID + 4];
  __shared__ float red[16][16];
  __shared__ float rowstat[16];
  const int lane = threadIdx.x & 31, wave = threadIdx.x >> 5;
  const int m0 = blockIdx.x * 16, n0 = wave * 16;
  v8f acc = {0.f, 0.f, 0.f, 0.f, 0.f, 0.f, 0.f, 0.f};
#pragma unroll
  for (int kc = 0; kc < DHID; kc += 32) {
    v16bf a = load_a_frag(Lb, N, m0, kc, lane);
    v16bf b = load_b_frag(Wab, n0, kc, lane);
    acc = __builtin_amdgcn_wmma_f32_16x16x32_bf16(false, a, false, b,
                                                  (short)0, acc, false, false);
  }
  const int nn = n0 + (lane & 15);
  const int mb = (lane >> 4) << 3;
#pragma unroll
  for (int r = 0; r < 8; ++r) ytile[mb + r][nn] = acc[r];
  __syncthreads();
  // fused row softmax (16 threads per row, 8 cols each)
  const int row = threadIdx.x >> 4, grp = threadIdx.x & 15;
  float* yr = ytile[row];
  float lmax = -3.0e38f;
  for (int c = grp * 8; c < grp * 8 + 8; ++c) lmax = fmaxf(lmax, yr[c]);
  red[row][grp] = lmax;
  __syncthreads();
  if (grp == 0) {
    float m = red[row][0];
    for (int i = 1; i < 16; ++i) m = fmaxf(m, red[row][i]);
    rowstat[row] = m;
  }
  __syncthreads();
  const float m = rowstat[row];
  float lsum = 0.f;
  for (int c = grp * 8; c < grp * 8 + 8; ++c) {
    float e = expf(yr[c] - m); yr[c] = e; lsum += e;
  }
  red[row][grp] = lsum;
  __syncthreads();
  if (grp == 0) {
    float s = 0.f;
    for (int i = 0; i < 16; ++i) s += red[row][i];
    rowstat[row] = s;
  }
  __syncthreads();
  const float inv = 1.0f / rowstat[row];
  const int mg = m0 + row;
  if (mg < N)
    for (int c = grp * 8; c < grp * 8 + 8; ++c)
      ass[(size_t)mg * DHID + c] = yr[c] * inv;
}

// y = Xb @ W^T, then Lorentz-linear epilogue -> out (q or k).
__global__ __launch_bounds__(256) void gemm_lorentz_kernel(const bf16_t* Xb,
                                                           const bf16_t* Wb,
                                                           const float* log_scale,
                                                           float* out, int N) {
  __shared__ float ytile[16][DHID + 4];
  __shared__ float red[16][16];
  __shared__ float rowstat[16];
  const int lane = threadIdx.x & 31, wave = threadIdx.x >> 5;
  const int m0 = blockIdx.x * 16, n0 = wave * 16;
  v8f acc = {0.f, 0.f, 0.f, 0.f, 0.f, 0.f, 0.f, 0.f};
#pragma unroll
  for (int kc = 0; kc < DHID; kc += 32) {
    v16bf a = load_a_frag(Xb, N, m0, kc, lane);
    v16bf b = load_b_frag(Wb, n0, kc, lane);
    acc = __builtin_amdgcn_wmma_f32_16x16x32_bf16(false, a, false, b,
                                                  (short)0, acc, false, false);
  }
  const int nn = n0 + (lane & 15);
  const int mb = (lane >> 4) << 3;
#pragma unroll
  for (int r = 0; r < 8; ++r) ytile[mb + r][nn] = acc[r];
  __syncthreads();
  const float scale = expf(log_scale[0]);
  const int row = threadIdx.x >> 4, grp = threadIdx.x & 15;
  float* yr = ytile[row];
  float lss = 0.f;
  for (int c = grp * 8; c < grp * 8 + 8; ++c)
    if (c > 0) lss += yr[c] * yr[c];
  red[row][grp] = lss;
  __syncthreads();
  if (grp == 0) {
    float s = 0.f;
    for (int i = 0; i < 16; ++i) s += red[row][i];
    rowstat[row] = s;
  }
  __syncthreads();
  const float sumsq = fmaxf(rowstat[row], EPSV);
  const float t = scale / (1.0f + expf(-yr[0])) + 1.1f;
  const float sq = sqrtf((t * t - 1.0f) / sumsq);
  const int mg = m0 + row;
  if (mg < N)
    for (int c = grp * 8; c < grp * 8 + 8; ++c)
      out[(size_t)mg * DHID + c] = (c == 0) ? t : yr[c] * sq;
}

__global__ __launch_bounds__(256) void init_kernel(float* acc, unsigned* segmax,
                                                   float* segsum, int N) {
  long i = (long)blockIdx.x * 256 + threadIdx.x;
  if (i < (long)N * DHID) acc[i] = 0.0f;
  if (i < N) { segmax[i] = 0x007FFFFFu; segsum[i] = 0.0f; }  // encoded(-inf)
}

// score[e] = -arccosh(clip(<q[src], k[dst]>_L)), atomic seg-max.  wave/edge.
__global__ __launch_bounds__(256) void edge_score_kernel(const float* q,
                                                         const float* k,
                                                         const int* src,
                                                         const int* dst,
                                                         float* score,
                                                         unsigned* segmax, int E) {
  const int lane = threadIdx.x & 31, wave = threadIdx.x >> 5;
  const long e = (long)blockIdx.x * 8 + wave;
  if (e >= E) return;
  const int s = src[e], d = dst[e];
  const float4 qa = ((const float4*)(q + (size_t)s * DHID))[lane];
  const float4 kb = ((const float4*)(k + (size_t)d * DHID))[lane];
  float dot = qa.x * kb.x + qa.y * kb.y + qa.z * kb.z + qa.w * kb.w;
  float p = (lane == 0) ? (2.0f * qa.x * kb.x - dot) : -dot;  // time term +
  for (int off = 16; off >= 1; off >>= 1) p += __shfl_xor(p, off, 32);
  if (lane == 0) {
    float inner = fmaxf(p, 1.0f + 1e-6f);
    float sc = -logf(inner + sqrtf(inner * inner - 1.0f));
    score[e] = sc;
    atomicMax(segmax + s, encodeOrd(sc));
  }
}

__global__ __launch_bounds__(256) void edge_exp_kernel(float* score,
                                                       const unsigned* segmax,
                                                       float* segsum,
                                                       const int* src, int E) {
  long e = (long)blockIdx.x * 256 + threadIdx.x;
  if (e >= E) return;
  const int s = src[e];
  float m = decodeOrd(segmax[s]);
  if (!(m > -3.0e38f)) m = 0.0f;
  const float e2 = expf(score[e] - m);
  score[e] = e2;                // in-place: score buffer now holds exp terms
  atomicAdd(segsum + s, e2);
}

// out_ass[src] += w[e] * ass[dst]  (102M f32 atomics, L2-resident)
__global__ __launch_bounds__(256) void edge_scatter_kernel(const float* e2,
                                                           const float* segsum,
                                                           const int* src,
                                                           const int* dst,
                                                           const float* ass,
                                                           float* acc, int E) {
  const int lane = threadIdx.x & 31, wave = threadIdx.x >> 5;
  const long e = (long)blockIdx.x * 8 + wave;
  if (e >= E) return;
  const int s = src[e], d = dst[e];
  float w = 0.f;
  if (lane == 0) w = e2[e] / fmaxf(segsum[s], EPSV);
  w = __shfl(w, 0, 32);
  const float4 a = ((const float4*)(ass + (size_t)d * DHID))[lane];
  float* o = acc + (size_t)s * DHID + lane * 4;
  atomicAdd(o + 0, w * a.x);
  atomicAdd(o + 1, w * a.y);
  atomicAdd(o + 2, w * a.z);
  atomicAdd(o + 3, w * a.w);
}

// logits + threefry-gumbel, row softmax.  wave per node row.
__global__ __launch_bounds__(256) void final_gumbel_kernel(const float* acc,
                                                           float* out, int N) {
  const int lane = threadIdx.x & 31, wave = threadIdx.x >> 5;
  const int row = blockIdx.x * 8 + wave;
  if (row >= N) return;
  const unsigned n = (unsigned)N * DHID, h = n >> 1;
  float vals[4];
#pragma unroll
  for (int j = 0; j < 4; ++j) {
    const unsigned i = (unsigned)row * DHID + lane * 4 + j;
    unsigned o0, o1, bits;
    if (i < h) { threefry2x32(0u, 42u, i, i + h, o0, o1); bits = o0; }
    else       { threefry2x32(0u, 42u, i - h, i, o0, o1); bits = o1; }
    float u01 = __uint_as_float((bits >> 9) | 0x3f800000u) - 1.0f;
    float u = fmaxf(1e-10f, u01 * (1.0f - 1e-10f) + 1e-10f);
    float g = -logf(-logf(u));
    float a = acc[(size_t)row * DHID + lane * 4 + j];
    vals[j] = (logf(a + 1e-6f) + g) * INV_TAU;
  }
  float lm = fmaxf(fmaxf(vals[0], vals[1]), fmaxf(vals[2], vals[3]));
  for (int off = 16; off >= 1; off >>= 1) lm = fmaxf(lm, __shfl_xor(lm, off, 32));
  float lsum = 0.f;
#pragma unroll
  for (int j = 0; j < 4; ++j) { vals[j] = expf(vals[j] - lm); lsum += vals[j]; }
  for (int off = 16; off >= 1; off >>= 1) lsum += __shfl_xor(lsum, off, 32);
  const float inv = 1.0f / lsum;
  ((float4*)(out + (size_t)row * DHID))[lane] =
      make_float4(vals[0] * inv, vals[1] * inv, vals[2] * inv, vals[3] * inv);
}

// ---------------------------------------------------------------- launch
extern "C" void kernel_launch(void* const* d_in, const int* in_sizes, int n_in,
                              void* d_out, int out_size, void* d_ws, size_t ws_size,
                              hipStream_t stream) {
  (void)n_in; (void)out_size; (void)ws_size;
  const float* x   = (const float*)d_in[0];
  const int*   src = (const int*)d_in[1];
  const int*   dst = (const int*)d_in[2];
  /* d_in[3] edge_value: unused by the reference */
  const float* Wq  = (const float*)d_in[4];
  const float* Wk  = (const float*)d_in[5];
  const float* Wa  = (const float*)d_in[6];
  const float* lsq = (const float*)d_in[7];
  const float* lsk = (const float*)d_in[8];
  const int N = in_sizes[0] / DHID;
  const int E = in_sizes[1];

  char* w = (char*)d_ws;
  size_t off = 0;
  auto take = [&](size_t bytes) -> size_t {
    size_t c = off; off += (bytes + 255) & ~(size_t)255; return c;
  };
  bf16_t*   Xb     = (bf16_t*)(w + take((size_t)N * DHID * sizeof(bf16_t)));
  bf16_t*   Lb     = (bf16_t*)(w + take((size_t)N * DHID * sizeof(bf16_t)));
  bf16_t*   Wqb    = (bf16_t*)(w + take((size_t)DHID * DHID * sizeof(bf16_t)));
  bf16_t*   Wkb    = (bf16_t*)(w + take((size_t)DHID * DHID * sizeof(bf16_t)));
  bf16_t*   Wab    = (bf16_t*)(w + take((size_t)DHID * DHID * sizeof(bf16_t)));
  float*    qv     = (float*)(w + take((size_t)N * DHID * sizeof(float)));
  float*    kv     = (float*)(w + take((size_t)N * DHID * sizeof(float)));
  float*    assv   = (float*)(w + take((size_t)N * DHID * sizeof(float)));
  float*    accv   = (float*)(w + take((size_t)N * DHID * sizeof(float)));
  float*    score  = (float*)(w + take((size_t)E * sizeof(float)));
  unsigned* segmax = (unsigned*)(w + take((size_t)N * sizeof(unsigned)));
  float*    segsum = (float*)(w + take((size_t)N * sizeof(float)));

  const int WELEM = DHID * DHID;
  convert_bf16_kernel<<<(WELEM + 255) / 256, 256, 0, stream>>>(Wq, Wqb, WELEM);
  convert_bf16_kernel<<<(WELEM + 255) / 256, 256, 0, stream>>>(Wk, Wkb, WELEM);
  convert_bf16_kernel<<<(WELEM + 255) / 256, 256, 0, stream>>>(Wa, Wab, WELEM);
  prep_nodes_kernel<<<(N + 7) / 8, 256, 0, stream>>>(x, Xb, Lb, N);

  const int gblk = (N + 15) / 16;
  gemm_softmax_kernel<<<gblk, 256, 0, stream>>>(Lb, Wab, assv, N);
  gemm_lorentz_kernel<<<gblk, 256, 0, stream>>>(Xb, Wqb, lsq, qv, N);
  gemm_lorentz_kernel<<<gblk, 256, 0, stream>>>(Xb, Wkb, lsk, kv, N);

  init_kernel<<<((long)N * DHID + 255) / 256, 256, 0, stream>>>(accv, segmax, segsum, N);
  edge_score_kernel<<<(E + 7) / 8, 256, 0, stream>>>(qv, kv, src, dst, score, segmax, E);
  edge_exp_kernel<<<(E + 255) / 256, 256, 0, stream>>>(score, segmax, segsum, src, E);
  edge_scatter_kernel<<<(E + 7) / 8, 256, 0, stream>>>(score, segsum, src, dst, assv, accv, E);
  final_gumbel_kernel<<<(N + 7) / 8, 256, 0, stream>>>(accv, (float*)d_out, N);
}
